// Transformer_Block_moe_43069932044303
// MI455X (gfx1250) — matrix operand order repaired and checked by reading, not compiled, not run on hardware
//
#include <hip/hip_runtime.h>

// ---------------------------------------------------------------------------
// Problem constants (from the reference)
// ---------------------------------------------------------------------------
#define B_    8
#define L_    1024
#define C_    1024
#define H_    16
#define DK_   64
#define E_    16
#define HID_  64
#define NTOK  (B_ * L_)      // 8192 tokens
#define CAP_  256            // 0.5 * 8192 / 16

typedef __attribute__((ext_vector_type(16))) __bf16 v16bf;
typedef __attribute__((ext_vector_type(8)))  __bf16 v8bf;
typedef __attribute__((ext_vector_type(8)))  float  v8f;

static __device__ inline v16bf cat16(v8bf lo, v8bf hi) {
  return __builtin_shufflevector(lo, hi, 0,1,2,3,4,5,6,7,8,9,10,11,12,13,14,15);
}

static __device__ inline v8f wmma_bf16(v16bf a, v16bf b, v8f c) {
  // D(f32 16x16) = A(bf16 16x32) * B(bf16 32x16) + C
  return __builtin_amdgcn_wmma_f32_16x16x32_bf16(false, a, false, b, (short)0, c,
                                                 false, false);
}

static __device__ inline v8f zero8f() {
  v8f z = {0.f,0.f,0.f,0.f,0.f,0.f,0.f,0.f};
  return z;
}

// ---------------------------------------------------------------------------
// Async global->LDS staging (gfx1250 GLOBAL_LOAD_ASYNC_TO_LDS_B128, ASYNCcnt).
// The builtin's pointer params are 'int __attribute__((vector_size(16)))*'
// (per compiler diagnostic); we build AS1/AS3-qualified pointers of exactly
// that pointee type. Low 32 bits of a generic LDS pointer are the LDS offset
// (ISA 10.2), so the AS3 pointer is built from the truncated address.
// ---------------------------------------------------------------------------
#if defined(__has_builtin)
#  if __has_builtin(__builtin_amdgcn_global_load_async_to_lds_b128) && \
      __has_builtin(__builtin_amdgcn_s_wait_asynccnt)
#    define USE_ASYNC_LDS 1
#  endif
#endif

typedef int v4i_t __attribute__((vector_size(4 * sizeof(int))));
typedef __attribute__((address_space(1))) v4i_t g_v4i;   // global
typedef __attribute__((address_space(3))) v4i_t l_v4i;   // LDS

static __device__ inline void cp16(__bf16* dst /*LDS*/, const __bf16* src /*global*/) {
#ifdef USE_ASYNC_LDS
  __builtin_amdgcn_global_load_async_to_lds_b128(
      (g_v4i*)(size_t)src,
      (l_v4i*)(size_t)(unsigned)(size_t)dst, 0, 0);
  __builtin_amdgcn_global_load_async_to_lds_b128(
      (g_v4i*)((size_t)src + 16),
      (l_v4i*)(size_t)((unsigned)(size_t)dst + 16), 0, 0);
#else
  *(v8bf*)dst       = *(const v8bf*)src;
  *(v8bf*)(dst + 8) = *(const v8bf*)(src + 8);
#endif
}

static __device__ inline void async_join() {
#ifdef USE_ASYNC_LDS
  __builtin_amdgcn_s_wait_asynccnt(0);
#endif
}

// ---------------------------------------------------------------------------
// One-time f32 -> bf16 conversion (activations and weights).
// ---------------------------------------------------------------------------
__global__ __launch_bounds__(256)
void cvt_bf16_kernel(const float* __restrict__ in, __bf16* __restrict__ out)
{
  const size_t i = ((size_t)blockIdx.x * 256 + threadIdx.x) * 4;
  const float4 v = *(const float4*)(in + i);
  out[i]     = (__bf16)v.x;
  out[i + 1] = (__bf16)v.y;
  out[i + 2] = (__bf16)v.z;
  out[i + 3] = (__bf16)v.w;
}

// W1 [E][C][HID] f32 -> W1t [E][HID][C] bf16  (N-major for WMMA B fragments)
__global__ __launch_bounds__(256)
void w1t_kernel(const float* __restrict__ W1, __bf16* __restrict__ W1t)
{
  __shared__ __bf16 T[64][72];
  const int tid = threadIdx.x;
  const int e  = blockIdx.x >> 4;
  const int c0 = (blockIdx.x & 15) * 64;
  const int r = tid >> 2, seg = (tid & 3) * 16;
  const float* src = W1 + ((size_t)e * C_ + c0 + r) * HID_ + seg;
#pragma unroll
  for (int i = 0; i < 16; ++i) T[seg + i][r] = (__bf16)src[i];
  __syncthreads();
  __bf16* dst = W1t + ((size_t)e * HID_ + r) * C_ + c0 + seg;
#pragma unroll
  for (int i = 0; i < 16; ++i) dst[i] = T[r][seg + i];
}

// W2 [E][HID][C] f32 -> W2t [E][C][HID] bf16
__global__ __launch_bounds__(256)
void w2t_kernel(const float* __restrict__ W2, __bf16* __restrict__ W2t)
{
  __shared__ __bf16 T[64][72];
  const int tid = threadIdx.x;
  const int e  = blockIdx.x >> 4;
  const int c0 = (blockIdx.x & 15) * 64;
  const int r = tid >> 2, seg = (tid & 3) * 16;
  const float* src = W2 + ((size_t)e * HID_ + r) * C_ + c0 + seg;
#pragma unroll
  for (int i = 0; i < 16; ++i) T[seg + i][r] = (__bf16)src[i];
  __syncthreads();
  __bf16* dst = W2t + ((size_t)e * C_ + c0 + r) * HID_ + seg;
#pragma unroll
  for (int i = 0; i < 16; ++i) dst[i] = T[r][seg + i];
}

// vb [NTOK][H*DK] bf16 -> vt [B*H][DK][L] bf16 (one-time transpose so the
// attention P@V B-fragments are contiguous row reads)
__global__ __launch_bounds__(256)
void vtrans_kernel(const __bf16* __restrict__ vb, __bf16* __restrict__ vt)
{
  __shared__ __bf16 T[64][72];
  const int tid = threadIdx.x;
  const int bh = blockIdx.x >> 4;          // 0..127
  const int l0 = (blockIdx.x & 15) * 64;
  const int b = bh >> 4, h = bh & 15;
  const int r = tid >> 2, seg = (tid & 3) * 16;
  const __bf16* src = vb + ((size_t)(b * L_ + l0 + r)) * (H_ * DK_) + h * DK_ + seg;
  v8bf lo = *(const v8bf*)src, hi = *(const v8bf*)(src + 8);
#pragma unroll
  for (int i = 0; i < 8; ++i) { T[seg + i][r] = lo[i]; T[seg + 8 + i][r] = hi[i]; }
  __syncthreads();
  __bf16* dst = vt + ((size_t)bh * DK_ + r) * L_ + l0 + seg;
  *(v8bf*)dst       = *(const v8bf*)&T[r][seg];
  *(v8bf*)(dst + 8) = *(const v8bf*)&T[r][seg + 8];
}

// ---------------------------------------------------------------------------
// Tiled bf16 GEMM:  Out[M,N] = scale * (A[M,K] @ W[N,K]^T + bias[N])
// Block: 256 threads (8 waves). Tile 128x128, K-chunk 64 (2 WMMA k-steps).
// ---------------------------------------------------------------------------
template <bool OUT_BF16>
__global__ __launch_bounds__(256)
void gemm_bf16_kernel(const __bf16* __restrict__ A, const __bf16* __restrict__ W,
                      const float* __restrict__ bias, void* __restrict__ Outv,
                      int M, int N, int K, float scale)
{
  __shared__ __bf16 As[128][72];   // [m][k], pitch 72 for bank spread
  __shared__ __bf16 Bs[128][72];   // [n][k] (N-major == W row-major)

  const int tid  = threadIdx.x;
  const int lane = tid & 31;
  const int wv   = tid >> 5;
  const int half = lane >> 4;
  const int l15  = lane & 15;
  const int m0 = blockIdx.y * 128;
  const int n0 = blockIdx.x * 128;
  const int wm = wv & 3;           // 32-row group
  const int wn = wv >> 2;          // 64-col group

  v8f acc[2][4];
#pragma unroll
  for (int i = 0; i < 2; ++i)
#pragma unroll
    for (int j = 0; j < 4; ++j) acc[i][j] = zero8f();

  const int srow = tid >> 1;            // 0..127
  const int sseg = (tid & 1) * 32;      // 0 or 32

  for (int k0 = 0; k0 < K; k0 += 64) {
    {
      const __bf16* sa = A + (size_t)(m0 + srow) * K + k0 + sseg;
      cp16(&As[srow][sseg],      sa);
      cp16(&As[srow][sseg + 16], sa + 16);
      const __bf16* sb = W + (size_t)(n0 + srow) * K + k0 + sseg;
      cp16(&Bs[srow][sseg],      sb);
      cp16(&Bs[srow][sseg + 16], sb + 16);
      if (k0 + 64 < K) __builtin_prefetch(sa + 64, 0, 3);
    }
    async_join();
    __syncthreads();

#pragma unroll
    for (int ks = 0; ks < 2; ++ks) {
      v16bf af[2], bfr[4];
#pragma unroll
      for (int mt = 0; mt < 2; ++mt) {
        const int r = 32 * wm + 16 * mt + l15;     // A: lane = M row
        af[mt] = cat16(*(const v8bf*)&As[r][32 * ks + 8 * half],
                       *(const v8bf*)&As[r][32 * ks + 16 + 8 * half]);
      }
#pragma unroll
      for (int nt = 0; nt < 4; ++nt) {
        const int n = 64 * wn + 16 * nt + l15;     // B: lane = N col
        bfr[nt] = cat16(*(const v8bf*)&Bs[n][32 * ks + 8 * half],
                        *(const v8bf*)&Bs[n][32 * ks + 16 + 8 * half]);
      }
#pragma unroll
      for (int mt = 0; mt < 2; ++mt)
#pragma unroll
        for (int nt = 0; nt < 4; ++nt)
          acc[mt][nt] = wmma_bf16(af[mt], bfr[nt], acc[mt][nt]);
    }
    __syncthreads();
  }

#pragma unroll
  for (int mt = 0; mt < 2; ++mt) {
#pragma unroll
    for (int nt = 0; nt < 4; ++nt) {
      const int col = n0 + 64 * wn + 16 * nt + l15;
      const float bv = bias[col];
#pragma unroll
      for (int r = 0; r < 8; ++r) {
        const int row = m0 + 32 * wm + 16 * mt + r + 8 * half;
        const float v = (acc[mt][nt][r] + bv) * scale;
        if (OUT_BF16) ((__bf16*)Outv)[(size_t)row * N + col] = (__bf16)v;
        else          ((float*)Outv)[(size_t)row * N + col]  = v;
      }
    }
  }
}

// ---------------------------------------------------------------------------
// Flash-style attention. One block (128 thr = 4 waves) per (b, h, 64-q-rows).
// Q frags pinned in registers; K staged row-major, V from pre-transposed vt.
// ---------------------------------------------------------------------------
__global__ __launch_bounds__(128)
void attn_kernel(const __bf16* __restrict__ qb, const __bf16* __restrict__ kb,
                 const __bf16* __restrict__ vt, __bf16* __restrict__ zb)
{
  __shared__ __bf16 Ks[32][72];       // [key][d]
  __shared__ __bf16 Vt[64][40];       // [d][key]
  __shared__ float  Sw[4][16][36];    // per-wave S tile 16x32
  __shared__ __bf16 Pw[4][16][40];    // per-wave P tile 16x32
  __shared__ float  factw[4][16];     // per-wave row factors

  const int tid  = threadIdx.x;
  const int lane = tid & 31;
  const int wv   = tid >> 5;
  const int half = lane >> 4;
  const int l15  = lane & 15;

  const int qblk = blockIdx.x & 15;        // L/64 = 16
  const int bh   = blockIdx.x >> 4;        // 0..127
  const int b    = bh >> 4;
  const int h    = bh & 15;
  const size_t seqbase = (size_t)b * L_;
  const int qr0 = qblk * 64 + wv * 16;

  v16bf aQ[2];
  {
    const __bf16* qrow = qb + (seqbase + qr0 + l15) * (H_ * DK_) + h * DK_;
#pragma unroll
    for (int ks = 0; ks < 2; ++ks)
      aQ[ks] = cat16(*(const v8bf*)(qrow + 32 * ks + 8 * half),
                     *(const v8bf*)(qrow + 32 * ks + 16 + 8 * half));
  }

  v8f accO[4];
#pragma unroll
  for (int nt = 0; nt < 4; ++nt) accO[nt] = zero8f();
  float mrow = -1e30f, lrow = 0.f;

  const int krow = tid >> 2;             // 0..31 (key row for K staging)
  const int kseg = (tid & 3) * 16;       // 0..48 (d segment)
  const int vrow = tid >> 1;             // 0..63 (d row for V staging)
  const int vseg = (tid & 1) * 16;       // 0 or 16 (key segment)

  for (int kk = 0; kk < L_ / 32; ++kk) {
    const int kbase = kk * 32;
    cp16(&Ks[krow][kseg],
         kb + (seqbase + kbase + krow) * (H_ * DK_) + h * DK_ + kseg);
    cp16(&Vt[vrow][vseg],
         vt + ((size_t)bh * DK_ + vrow) * L_ + kbase + vseg);
    async_join();
    __syncthreads();

    // ---- S(16x32) = Q(16x64) @ K^T ----
    v8f accS[2];
    accS[0] = zero8f(); accS[1] = zero8f();
#pragma unroll
    for (int nt = 0; nt < 2; ++nt)
#pragma unroll
      for (int ks = 0; ks < 2; ++ks) {
        const __bf16* kp = &Ks[16 * nt + l15][32 * ks + 8 * half];
        accS[nt] = wmma_bf16(aQ[ks], cat16(*(const v8bf*)kp, *(const v8bf*)(kp + 16)),
                             accS[nt]);
      }
#pragma unroll
    for (int nt = 0; nt < 2; ++nt)
#pragma unroll
      for (int r = 0; r < 8; ++r)
        Sw[wv][r + 8 * half][16 * nt + l15] = accS[nt][r];

    // ---- online softmax, lanes 0..15 own rows ----
    if (lane < 16) {
      float sv[32];
      float bmax = -1e30f;
#pragma unroll
      for (int j = 0; j < 32; ++j) { sv[j] = Sw[wv][lane][j]; bmax = fmaxf(bmax, sv[j]); }
      const float newm  = fmaxf(mrow, bmax);
      const float alpha = __expf(mrow - newm);
      float psum = 0.f;
#pragma unroll
      for (int j = 0; j < 32; ++j) {
        const float p = __expf(sv[j] - newm);
        psum += p;
        Pw[wv][lane][j] = (__bf16)p;
      }
      lrow = lrow * alpha + psum;
      mrow = newm;
      factw[wv][lane] = alpha;
    }
#pragma unroll
    for (int nt = 0; nt < 4; ++nt)
#pragma unroll
      for (int r = 0; r < 8; ++r)
        accO[nt][r] *= factw[wv][r + 8 * half];

    // ---- O(16x64) += P(16x32) @ V(32x64) ----
    v16bf aP;
    {
      const __bf16* pp = &Pw[wv][l15][0];
      aP = cat16(*(const v8bf*)(pp + 8 * half),
                 *(const v8bf*)(pp + 16 + 8 * half));
    }
#pragma unroll
    for (int nt = 0; nt < 4; ++nt) {
      const __bf16* vp = &Vt[16 * nt + l15][8 * half];
      accO[nt] = wmma_bf16(aP, cat16(*(const v8bf*)vp, *(const v8bf*)(vp + 16)),
                           accO[nt]);
    }
    __syncthreads();
  }

  if (lane < 16) factw[wv][lane] = 1.0f / lrow;
#pragma unroll
  for (int nt = 0; nt < 4; ++nt)
#pragma unroll
    for (int r = 0; r < 8; ++r) {
      const float inv = factw[wv][r + 8 * half];
      const size_t row = seqbase + qr0 + r + 8 * half;
      zb[row * (H_ * DK_) + h * DK_ + 16 * nt + l15] = (__bf16)(accO[nt][r] * inv);
    }
}

// ---------------------------------------------------------------------------
// zn = LayerNorm(qx + att) * w + b ; also bf16 copy for the MoE GEMMs.
// ---------------------------------------------------------------------------
__global__ __launch_bounds__(256)
void ln_kernel(const float* __restrict__ qx, const float* __restrict__ att,
               const float* __restrict__ lw, const float* __restrict__ lb,
               float* __restrict__ zn, __bf16* __restrict__ znb)
{
  __shared__ float red[256];
  const int row = blockIdx.x, tid = threadIdx.x;
  const float* q = qx  + (size_t)row * C_;
  const float* a = att + (size_t)row * C_;

  float x[4];
  float s = 0.f;
#pragma unroll
  for (int i = 0; i < 4; ++i) {
    const int c = tid + 256 * i;
    x[i] = q[c] + a[c];
    s += x[i];
  }
  red[tid] = s; __syncthreads();
  for (int off = 128; off; off >>= 1) {
    if (tid < off) red[tid] += red[tid + off];
    __syncthreads();
  }
  const float mu = red[0] * (1.f / C_);
  __syncthreads();

  float vs = 0.f;
#pragma unroll
  for (int i = 0; i < 4; ++i) { const float d = x[i] - mu; vs += d * d; }
  red[tid] = vs; __syncthreads();
  for (int off = 128; off; off >>= 1) {
    if (tid < off) red[tid] += red[tid + off];
    __syncthreads();
  }
  const float rstd = rsqrtf(red[0] * (1.f / C_) + 1e-5f);

#pragma unroll
  for (int i = 0; i < 4; ++i) {
    const int c = tid + 256 * i;
    const float y = (x[i] - mu) * rstd * lw[c] + lb[c];
    zn[(size_t)row * C_ + c]  = y;
    znb[(size_t)row * C_ + c] = (__bf16)y;
  }
}

// ---------------------------------------------------------------------------
// Router: routes[t] = argmax_e (zn[t] . Wsw[e] + bsw[e]).  (fwd scale == 1)
// ---------------------------------------------------------------------------
__global__ __launch_bounds__(256)
void router_kernel(const float* __restrict__ zn, const float* __restrict__ Wsw,
                   const float* __restrict__ bsw, int* __restrict__ routes)
{
  const int tid = threadIdx.x, lane = tid & 31, wv = tid >> 5;
  const int tok = blockIdx.x * 8 + wv;
  const float* x = zn + (size_t)tok * C_;

  float acc[E_];
#pragma unroll
  for (int e = 0; e < E_; ++e) acc[e] = 0.f;
  for (int i = 0; i < 32; ++i) {
    const int c = lane + 32 * i;
    const float xv = x[c];
#pragma unroll
    for (int e = 0; e < E_; ++e) acc[e] += xv * Wsw[e * C_ + c];
  }
#pragma unroll
  for (int e = 0; e < E_; ++e) {
#pragma unroll
    for (int off = 16; off; off >>= 1) acc[e] += __shfl_xor(acc[e], off, 32);
  }
  if (lane == 0) {
    int best = 0;
    float bv = acc[0] + bsw[0];
#pragma unroll
    for (int e = 1; e < E_; ++e) {
      const float v = acc[e] + bsw[e];
      if (v > bv) { bv = v; best = e; }   // first max wins (matches argmax)
    }
    routes[tok] = best;
  }
}

// ---------------------------------------------------------------------------
// Deterministic capacity scan (single block).
// ---------------------------------------------------------------------------
__global__ __launch_bounds__(256)
void route_scan_kernel(const int* __restrict__ routes, int* __restrict__ etok)
{
  __shared__ int scanb[256];
  __shared__ int basel[256][E_];
  const int tid = threadIdx.x;

  for (int i = tid; i < E_ * CAP_; i += 256) etok[i] = -1;

  int loc[E_];
#pragma unroll
  for (int e = 0; e < E_; ++e) loc[e] = 0;
  for (int i = 0; i < 32; ++i) loc[routes[tid * 32 + i]]++;
  __syncthreads();

  for (int e = 0; e < E_; ++e) {
    const int v = loc[e];
    scanb[tid] = v; __syncthreads();
    for (int off = 1; off < 256; off <<= 1) {
      const int add = (tid >= off) ? scanb[tid - off] : 0;
      __syncthreads();
      scanb[tid] += add;
      __syncthreads();
    }
    basel[tid][e] = scanb[tid] - v;
    __syncthreads();
  }

  int off_[E_];
#pragma unroll
  for (int e = 0; e < E_; ++e) off_[e] = basel[tid][e];
  for (int i = 0; i < 32; ++i) {
    const int t = tid * 32 + i;
    const int e = routes[t];
    const int pos = off_[e]++;
    if (pos < CAP_) etok[e * CAP_ + pos] = t;
  }
}

__global__ __launch_bounds__(256)
void init_out_kernel(const float* __restrict__ zn, float* __restrict__ out)
{
  const size_t i = (size_t)blockIdx.x * 1024 + (size_t)threadIdx.x * 4;
  const float4 v = *(const float4*)(zn + i);
  float4 o; o.x = 2.f*v.x; o.y = 2.f*v.y; o.z = 2.f*v.z; o.w = 2.f*v.w;
  *(float4*)(out + i) = o;
}

__global__ __launch_bounds__(256)
void mask_kernel(const unsigned char* __restrict__ m, float* __restrict__ out)
{
  const int i = blockIdx.x * 256 + threadIdx.x;
  out[i] = m[i] ? 1.0f : 0.0f;
}

// ---------------------------------------------------------------------------
// MoE FFN. Block per (expert, 64-row slab). Weights pre-transposed to bf16.
// ---------------------------------------------------------------------------
__global__ __launch_bounds__(256)
void moe_kernel(const __bf16* __restrict__ znb, const float* __restrict__ zn,
                const __bf16* __restrict__ W1t, const float* __restrict__ b1,
                const __bf16* __restrict__ W2t, const float* __restrict__ b2,
                const int* __restrict__ etok, float* __restrict__ out)
{
  __shared__ __bf16 Xs [64][72];     // gathered activations [row][c_local]
  __shared__ __bf16 W1s[64][72];     // W1t chunk [h][c_local]
  __shared__ __bf16 Hs [64][72];     // hidden (post-relu)   [row][h]
  __shared__ __bf16 W2s[128][72];    // W2t chunk [c_local][h]
  __shared__ int    toks[64];

  const int tid  = threadIdx.x;
  const int lane = tid & 31;
  const int wv   = tid >> 5;
  const int half = lane >> 4;
  const int l15  = lane & 15;
  const int e    = blockIdx.x >> 2;
  const int r0   = (blockIdx.x & 3) * 64;

  if (tid < 64) toks[tid] = etok[e * CAP_ + r0 + tid];
  __syncthreads();

  const int rtile = wv & 3;
  const int nh    = wv >> 2;

  const __bf16* W1e = W1t + (size_t)e * HID_ * C_;   // [HID][C]
  const float*  b1e = b1  + (size_t)e * HID_;
  const __bf16* W2e = W2t + (size_t)e * C_ * HID_;   // [C][HID]
  const float*  b2e = b2  + (size_t)e * C_;

  const int srow = tid >> 2;        // 0..63
  const int sseg = (tid & 3) * 16;  // 0..48

  // ---- GEMM1: H(64x64) = X(64x1024) @ W1(1024x64) ----
  v8f acc1[2];
  acc1[0] = zero8f(); acc1[1] = zero8f();
  for (int c0 = 0; c0 < C_; c0 += 64) {
    {
      const int tok = toks[srow];
      if (tok >= 0) {
        cp16(&Xs[srow][sseg], znb + (size_t)tok * C_ + c0 + sseg);
      } else {
#pragma unroll
        for (int i = 0; i < 16; ++i) Xs[srow][sseg + i] = (__bf16)0.f;
      }
      cp16(&W1s[srow][sseg], W1e + (size_t)srow * C_ + c0 + sseg);
    }
    async_join();
    __syncthreads();
#pragma unroll
    for (int ks = 0; ks < 2; ++ks) {
      const int ar = 16 * rtile + l15;
      v16bf aX = cat16(*(const v8bf*)&Xs[ar][32 * ks + 8 * half],
                       *(const v8bf*)&Xs[ar][32 * ks + 16 + 8 * half]);
#pragma unroll
      for (int nt = 0; nt < 2; ++nt) {
        const int hr = 32 * nh + 16 * nt + l15;
        v16bf bW = cat16(*(const v8bf*)&W1s[hr][32 * ks + 8 * half],
                         *(const v8bf*)&W1s[hr][32 * ks + 16 + 8 * half]);
        acc1[nt] = wmma_bf16(aX, bW, acc1[nt]);
      }
    }
    __syncthreads();
  }

  // bias + relu -> Hs (bf16)
#pragma unroll
  for (int nt = 0; nt < 2; ++nt) {
    const int hcol = 32 * nh + 16 * nt + l15;
    const float bv = b1e[hcol];
#pragma unroll
    for (int r = 0; r < 8; ++r) {
      float v = acc1[nt][r] + bv;
      v = v > 0.f ? v : 0.f;
      Hs[16 * rtile + r + 8 * half][hcol] = (__bf16)v;
    }
  }
  __syncthreads();

  // ---- GEMM2: Y(64x1024) = H(64x64) @ W2(64x1024), 128 cols per chunk ----
  const int crow = tid >> 1;           // 0..127 (c row of W2s)
  const int cseg = (tid & 1) * 32;     // 0 or 32 (h segment)
  for (int cb = 0; cb < C_; cb += 128) {
    {
      const __bf16* src = W2e + (size_t)(cb + crow) * HID_ + cseg;
      cp16(&W2s[crow][cseg],      src);
      cp16(&W2s[crow][cseg + 16], src + 16);
    }
    async_join();
    __syncthreads();

    v8f acc2[4];
#pragma unroll
    for (int nt = 0; nt < 4; ++nt) acc2[nt] = zero8f();
#pragma unroll
    for (int ks = 0; ks < 2; ++ks) {
      const int ar = 16 * rtile + l15;
      v16bf aH = cat16(*(const v8bf*)&Hs[ar][32 * ks + 8 * half],
                       *(const v8bf*)&Hs[ar][32 * ks + 16 + 8 * half]);
#pragma unroll
      for (int nt = 0; nt < 4; ++nt) {
        const int nr = 64 * nh + 16 * nt + l15;
        v16bf bW = cat16(*(const v8bf*)&W2s[nr][32 * ks + 8 * half],
                         *(const v8bf*)&W2s[nr][32 * ks + 16 + 8 * half]);
        acc2[nt] = wmma_bf16(aH, bW, acc2[nt]);
      }
    }
#pragma unroll
    for (int nt = 0; nt < 4; ++nt) {
      const int col = cb + 64 * nh + 16 * nt + l15;
      const float bv = b2e[col];
#pragma unroll
      for (int r = 0; r < 8; ++r) {
        const int row = 16 * rtile + r + 8 * half;
        const int tok = toks[row];
        if (tok >= 0) {
          const size_t idx = (size_t)tok * C_ + col;
          out[idx] = zn[idx] + acc2[nt][r] + bv;
        }
      }
    }
    __syncthreads();
  }
}

// ---------------------------------------------------------------------------
// Host launcher
// ---------------------------------------------------------------------------
extern "C" void kernel_launch(void* const* d_in, const int* in_sizes, int n_in,
                              void* d_out, int out_size, void* d_ws, size_t ws_size,
                              hipStream_t stream)
{
  (void)in_sizes; (void)n_in; (void)out_size; (void)ws_size;

  const float* qx   = (const float*)d_in[0];
  const float* kx   = (const float*)d_in[1];
  const float* vx   = (const float*)d_in[2];
  const unsigned char* maskPAD = (const unsigned char*)d_in[3];
  const float* WQ = (const float*)d_in[4];
  const float* bQ = (const float*)d_in[5];
  const float* WK = (const float*)d_in[6];
  const float* bK = (const float*)d_in[7];
  const float* WV = (const float*)d_in[8];
  const float* bV = (const float*)d_in[9];
  const float* WO = (const float*)d_in[10];
  const float* bO = (const float*)d_in[11];
  const float* ln1w = (const float*)d_in[12];
  const float* ln1b = (const float*)d_in[13];
  const float* Wsw  = (const float*)d_in[14];
  const float* bsw  = (const float*)d_in[15];
  const float* W1   = (const float*)d_in[16];
  const float* b1   = (const float*)d_in[17];
  const float* W2   = (const float*)d_in[18];
  const float* b2   = (const float*)d_in[19];

  const size_t NC = (size_t)NTOK * C_;      // 8388608
  const size_t CC = (size_t)C_ * C_;        // 1048576
  float* out      = (float*)d_out;
  float* out_kx   = out + NC;
  float* out_vx   = out_kx + NC;
  float* out_mask = out_vx + NC;

  // workspace layout (MB16 = one bf16 [NTOK,C] panel)
  char* ws = (char*)d_ws;
  const size_t MB16 = NC * 2;               // 16 MiB
  const size_t MB2  = CC * 2;               // 2 MiB
  __bf16* qxb = (__bf16*)(ws);                       // later reused as vt
  __bf16* kxb = (__bf16*)(ws + 1 * MB16);            // later reused as zb
  __bf16* vxb = (__bf16*)(ws + 2 * MB16);            // later reused as znb
  __bf16* qb  = (__bf16*)(ws + 3 * MB16);
  __bf16* kb  = (__bf16*)(ws + 4 * MB16);
  __bf16* vb  = (__bf16*)(ws + 5 * MB16);
  __bf16* WQb = (__bf16*)(ws + 6 * MB16);
  __bf16* WKb = (__bf16*)(ws + 6 * MB16 + 1 * MB2);
  __bf16* WVb = (__bf16*)(ws + 6 * MB16 + 2 * MB2);
  __bf16* WOb = (__bf16*)(ws + 6 * MB16 + 3 * MB2);
  __bf16* W1tb= (__bf16*)(ws + 6 * MB16 + 4 * MB2);
  __bf16* W2tb= (__bf16*)(ws + 6 * MB16 + 5 * MB2);
  float*  zn  = (float*) (ws + 6 * MB16 + 6 * MB2);  // 32 MiB
  int* routes = (int*)   (ws + 8 * MB16 + 6 * MB2);
  int* etok   = (int*)   (ws + 8 * MB16 + 6 * MB2 + 65536);

  __bf16* vt  = qxb;   // safe: qxb consumed by Q-GEMM before vtrans runs
  __bf16* zb  = kxb;   // safe: kxb consumed by K-GEMM before attention runs
  __bf16* znb = vxb;   // safe: vxb consumed by V-GEMM before LN runs
  float*  att = out;   // scratch aliases output region; consumed before overwrite

  const dim3 blk256(256), blk128(128);
  const dim3 gGemm(C_ / 128, NTOK / 128);  // (8, 64)

  // one-time bf16 conversions + weight transposes
  cvt_bf16_kernel<<<dim3(NC / 1024), blk256, 0, stream>>>(qx, qxb);
  cvt_bf16_kernel<<<dim3(NC / 1024), blk256, 0, stream>>>(kx, kxb);
  cvt_bf16_kernel<<<dim3(NC / 1024), blk256, 0, stream>>>(vx, vxb);
  cvt_bf16_kernel<<<dim3(CC / 1024), blk256, 0, stream>>>(WQ, WQb);
  cvt_bf16_kernel<<<dim3(CC / 1024), blk256, 0, stream>>>(WK, WKb);
  cvt_bf16_kernel<<<dim3(CC / 1024), blk256, 0, stream>>>(WV, WVb);
  cvt_bf16_kernel<<<dim3(CC / 1024), blk256, 0, stream>>>(WO, WOb);
  w1t_kernel<<<dim3(E_ * 16), blk256, 0, stream>>>(W1, W1tb);
  w2t_kernel<<<dim3(E_ * 16), blk256, 0, stream>>>(W2, W2tb);

  // Q/K/V projections (bf16 out; 1/sqrt(64) folded into Q)
  gemm_bf16_kernel<true><<<gGemm, blk256, 0, stream>>>(qxb, WQb, bQ, qb, NTOK, C_, C_, 0.125f);
  gemm_bf16_kernel<true><<<gGemm, blk256, 0, stream>>>(kxb, WKb, bK, kb, NTOK, C_, C_, 1.0f);
  gemm_bf16_kernel<true><<<gGemm, blk256, 0, stream>>>(vxb, WVb, bV, vb, NTOK, C_, C_, 1.0f);

  // one-time V transpose for attention B fragments
  vtrans_kernel<<<dim3(B_ * H_ * 16), blk256, 0, stream>>>(vb, vt);

  // attention
  attn_kernel<<<dim3(B_ * H_ * (L_ / 64)), blk128, 0, stream>>>(qb, kb, vt, zb);

  // output projection (f32 out into att scratch)
  gemm_bf16_kernel<false><<<gGemm, blk256, 0, stream>>>(zb, WOb, bO, att, NTOK, C_, C_, 1.0f);

  // residual + layernorm
  ln_kernel<<<dim3(NTOK), blk256, 0, stream>>>(qx, att, ln1w, ln1b, zn, znb);

  // router + deterministic capacity scan
  router_kernel<<<dim3(NTOK / 8), blk256, 0, stream>>>(zn, Wsw, bsw, routes);
  route_scan_kernel<<<dim3(1), blk256, 0, stream>>>(routes, etok);

  // out = 2*zn baseline (dropped tokens), then MoE overwrites kept tokens
  init_out_kernel<<<dim3(NTOK), blk256, 0, stream>>>(zn, out);
  moe_kernel<<<dim3(E_ * 4), blk256, 0, stream>>>(znb, zn, W1tb, b1, W2tb, b2, etok, out);

  // passthrough outputs (kx, vx, maskPAD)
  (void)hipMemcpyAsync(out_kx, kx, NC * sizeof(float), hipMemcpyDeviceToDevice, stream);
  (void)hipMemcpyAsync(out_vx, vx, NC * sizeof(float), hipMemcpyDeviceToDevice, stream);
  mask_kernel<<<dim3(NTOK / 256), blk256, 0, stream>>>(maskPAD, out_mask);
}
